// GCN_7129645711835
// MI455X (gfx1250) — matrix operand order repaired and checked by reading, not compiled, not run on hardware
//
#include <hip/hip_runtime.h>

typedef __attribute__((ext_vector_type(16))) _Float16 v16h;
typedef __attribute__((ext_vector_type(8)))  _Float16 v8h;
typedef __attribute__((ext_vector_type(8)))  float    v8f;
typedef __attribute__((ext_vector_type(4)))  unsigned v4u;
typedef __attribute__((ext_vector_type(8)))  int      v8i;
typedef __attribute__((ext_vector_type(4)))  int      v4i;

#define BN_EPS 1e-5f

#if defined(__has_builtin)
#if __has_builtin(__builtin_amdgcn_tensor_load_to_lds) && (__clang_major__ < 23)
#define USE_TDM 1
#endif
#endif

// ---------------- utility kernels ----------------

__global__ void fill_f32(float* __restrict__ p, float v, unsigned n) {
    unsigned i = blockIdx.x * blockDim.x + threadIdx.x;
    if (i < n) p[i] = v;
}

__global__ void deg_count(const int* __restrict__ dst, float* __restrict__ deg, unsigned E) {
    unsigned e = blockIdx.x * blockDim.x + threadIdx.x;
    if (e < E) atomicAdd(&deg[dst[e]], 1.0f);
}

__global__ void deg_to_dinv(float* __restrict__ deg, unsigned N) {
    unsigned i = blockIdx.x * blockDim.x + threadIdx.x;
    if (i < N) {
        float d = deg[i];
        deg[i] = (d > 0.0f) ? rsqrtf(d) : 0.0f;
    }
}

__global__ void f32_to_f16(const float* __restrict__ x, _Float16* __restrict__ y, unsigned n) {
    unsigned i = blockIdx.x * blockDim.x + threadIdx.x;
    if (i < n) y[i] = (_Float16)x[i];
}

// Pack row-major W[K, Ncols] into WMMA B-operand layout, zero-padding columns
// to NT*16.  Bp index = ((ks*NT + t)*32 + lane)*16 + i  maps to element
// (k = ks*32 + (lane/16)*16 + i,  n = t*16 + lane%16).
__global__ void pack_B(const float* __restrict__ W, _Float16* __restrict__ Bp,
                       int K, int Ncols, int NT) {
    int idx = blockIdx.x * blockDim.x + threadIdx.x;
    int total = (K >> 5) * NT * 512;
    if (idx >= total) return;
    int i    = idx & 15;
    int lane = (idx >> 4) & 31;
    int rest = idx >> 9;          // = ks*NT + t
    int t    = rest % NT;
    int ks   = rest / NT;
    int k = ks * 32 + (lane >> 4) * 16 + i;
    int n = t * 16 + (lane & 15);
    float v = (n < Ncols) ? W[(size_t)k * Ncols + n] : 0.0f;
    Bp[idx] = (_Float16)v;
}

#ifdef USE_TDM
// Issue one Tensor-Data-Mover copy of `ndwords` DWORDs from global -> LDS.
// D# built per CDNA5 ISA (1-D tile, data_size=4B, type=2).
__device__ static inline void tdm_load_1d(const void* gptr, void* lptr, unsigned ndwords) {
    unsigned lds = (unsigned)(unsigned long long)(uintptr_t)lptr;   // LDS byte offset
    unsigned long long ga = (unsigned long long)(uintptr_t)gptr;

    v4u g0;
    g0.x = 1u;                                   // count=1, user descriptor
    g0.y = lds;                                  // lds_addr
    g0.z = (unsigned)(ga & 0xFFFFFFFFu);         // global_addr[31:0]
    g0.w = ((unsigned)(ga >> 32) & 0x01FFFFFFu)  // global_addr[56:32]
           | 0x80000000u;                        // type=2 ("image") in [127:126]

    v8i g1;
    g1[0] = (int)(2u << 16);                     // data_size=2 -> 4 bytes
    g1[1] = (int)((ndwords & 0xFFFFu) << 16);    // tensor_dim0[15:0]
    g1[2] = (int)(((ndwords >> 16) & 0xFFFFu)    // tensor_dim0[31:16]
           | (1u << 16));                        // tensor_dim1 = 1
    g1[3] = (int)((ndwords & 0xFFFFu) << 16);    // tile_dim0 = ndwords
    g1[4] = (int)((1u) | (1u << 16));            // tile_dim1 = 1, tile_dim2 = 1
    g1[5] = (int)ndwords;                        // tensor_dim0_stride[31:0]
    g1[6] = 0;                                   // stride hi / dim1_stride lo
    g1[7] = 0;

    v4i g2;                                      // normal mode group 2
    g2.x = 1;                                    // tensor_dim2 = 1
    g2.y = 1;                                    // tensor_dim3 = 1
    g2.z = 0;                                    // tensor_dim2_stride lo
    g2.w = (int)(1u << 16);                      // tile_dim3 = 1

    v4i g3;                                      // normal mode group 3
    g3.x = 0;                                    // tensor_dim3_stride lo
    g3.y = (int)(1u << 16);                      // tensor_dim4 = 1
    g3.z = (int)(1u << 16);                      // tile_dim4 = 1
    g3.w = 0;

    __builtin_amdgcn_tensor_load_to_lds(g0, g1, g2, g3, 0 /*cpol*/);
}
#endif

// ---------------- WMMA GEMM:  C[M, writeCols] = A[M, K] * Bp ----------------
// Packed B is staged in LDS once per workgroup (TDM if available), then each
// wave computes a 16-row x (NT*16)-col tile.  M must be a multiple of 16.
template <int NT>
__global__ void gemm_wmma(const _Float16* __restrict__ A,
                          const _Float16* __restrict__ Bp,
                          float* __restrict__ C,
                          int K, int numTiles, int writeCols) {
    __shared__ __align__(16) _Float16 Bs[8192];   // 16 KB, covers K=128 x 64 cols

    int lane      = threadIdx.x & 31;
    int waveInBlk = threadIdx.x >> 5;
    int wave      = blockIdx.x * (blockDim.x >> 5) + waveInBlk;

    unsigned ndw = (unsigned)(((K >> 5) * NT * 512) >> 1);   // halves -> dwords

#ifdef USE_TDM
    if (waveInBlk == 0) {                         // wave-uniform branch
        tdm_load_1d(Bp, Bs, ndw);
        __builtin_amdgcn_s_wait_tensorcnt(0);     // TDM completion, this wave
    }
#else
    for (unsigned t = threadIdx.x; t < ndw; t += blockDim.x)
        ((unsigned*)Bs)[t] = ((const unsigned*)Bp)[t];
#endif
    __syncthreads();                              // publish LDS to all waves

    if (wave < numTiles) {
        int m0    = wave * 16;
        int mrow  = m0 + (lane & 15);             // A: row owned by this lane
        int kbase = (lane >> 4) * 16;             // A: K sub-block per half-wave

        v8f c[NT] = {};                           // fp32 accumulators

        int ksteps = K >> 5;
        for (int ks = 0; ks < ksteps; ++ks) {
            int kk = ks << 5;
            const _Float16* ap = A + (size_t)mrow * K + kk + kbase;
            v8h alo = *(const v8h*)ap;            // K = kbase+0..7
            v8h ahi = *(const v8h*)(ap + 8);      // K = kbase+8..15
            v16h a;
#pragma unroll
            for (int i = 0; i < 8; ++i) { a[i] = alo[i]; a[i + 8] = ahi[i]; }

#pragma unroll
            for (int t = 0; t < NT; ++t) {
                const _Float16* bp = Bs + ((ks * NT + t) * 32 + lane) * 16;
                v8h blo = *(const v8h*)bp;        // ds_load_b128
                v8h bhi = *(const v8h*)(bp + 8);
                v16h b;
#pragma unroll
                for (int i = 0; i < 8; ++i) { b[i] = blo[i]; b[i + 8] = bhi[i]; }

                c[t] = __builtin_amdgcn_wmma_f32_16x16x32_f16(
                           false, a, false, b, (short)0, c[t], false, false);
            }
        }

        // D layout: lane L -> n = L%16 ; VGPR j -> m = (L/16)*8 + j
        int nb = lane & 15;
        int mh = (lane >> 4) * 8;
#pragma unroll
        for (int t = 0; t < NT; ++t) {
#pragma unroll
            for (int j = 0; j < 8; ++j) {
                int m = m0 + mh + j;
                int n = t * 16 + nb;
                if (n < writeCols)
                    C[(size_t)m * writeCols + n] = c[t][j];
            }
        }
    }
}

// ---------------- graph aggregation ----------------

// One thread per (edge, feature): out[dst,f] += h[src,f] * dinv[src]*dinv[dst]
__global__ void scatter_edges(const int* __restrict__ src, const int* __restrict__ dst,
                              const float* __restrict__ dinv, const float* __restrict__ h,
                              float* __restrict__ out, unsigned E, int F) {
    unsigned idx = blockIdx.x * blockDim.x + threadIdx.x;
    unsigned total = E * (unsigned)F;
    if (idx >= total) return;
    unsigned e = idx / (unsigned)F;
    unsigned f = idx - e * (unsigned)F;
    int s = src[e], d = dst[e];
    float nrm = dinv[s] * dinv[d];
    atomicAdd(&out[(size_t)d * F + f], h[(size_t)s * F + f] * nrm);
}

// self-loop term + bias:  out[i,f] += h[i,f]*dinv[i]^2 + b[f]
__global__ void post_agg(const float* __restrict__ h, const float* __restrict__ dinv,
                         const float* __restrict__ bias, float* __restrict__ out,
                         unsigned N, int F) {
    unsigned idx = blockIdx.x * blockDim.x + threadIdx.x;
    if (idx >= N * (unsigned)F) return;
    unsigned i = idx / (unsigned)F;
    unsigned f = idx - i * (unsigned)F;
    float dv = dinv[i];
    out[idx] += h[idx] * dv * dv + bias[f];
}

// per-feature sum / sum-of-squares over N rows (F = 64 fixed)
__global__ void bn_stats(const float* __restrict__ x, float* __restrict__ sums, unsigned N) {
    int f = threadIdx.x & 63;
    unsigned row = blockIdx.x * (blockDim.x >> 6) + (threadIdx.x >> 6);
    unsigned rstride = gridDim.x * (blockDim.x >> 6);
    float s = 0.0f, s2 = 0.0f;
    for (unsigned r = row; r < N; r += rstride) {
        float v = x[(size_t)r * 64 + f];
        s += v; s2 += v * v;
    }
    atomicAdd(&sums[f], s);
    atomicAdd(&sums[64 + f], s2);
}

// batchnorm + relu, emit f16 directly for the next WMMA GEMM (F = 64)
__global__ void bn_relu_f16(const float* __restrict__ x, const float* __restrict__ sums,
                            const float* __restrict__ gamma, const float* __restrict__ beta,
                            _Float16* __restrict__ y, unsigned N) {
    unsigned idx = blockIdx.x * blockDim.x + threadIdx.x;
    if (idx >= N * 64u) return;
    int f = idx & 63;
    float invN = 1.0f / (float)N;
    float mean = sums[f] * invN;
    float var  = sums[64 + f] * invN - mean * mean;
    float rstd = rsqrtf(var + BN_EPS);
    float v = gamma[f] * (x[idx] - mean) * rstd + beta[f];
    v = v > 0.0f ? v : 0.0f;
    y[idx] = (_Float16)v;
}

// ---------------- launcher ----------------

static inline unsigned cdiv(size_t a, size_t b) { return (unsigned)((a + b - 1) / b); }

extern "C" void kernel_launch(void* const* d_in, const int* in_sizes, int n_in,
                              void* d_out, int out_size, void* d_ws, size_t ws_size,
                              hipStream_t stream) {
    const float* node_feat = (const float*)d_in[0];
    const int*   edge      = (const int*)  d_in[1];
    const float* W1        = (const float*)d_in[2];
    const float* b1        = (const float*)d_in[3];
    const float* gamma     = (const float*)d_in[4];
    const float* beta      = (const float*)d_in[5];
    const float* W2        = (const float*)d_in[6];
    const float* b2        = (const float*)d_in[7];

    const int HID = in_sizes[3];                  // 64
    const int OUT = in_sizes[7];                  // 40
    const int IN  = in_sizes[2] / HID;            // 128
    const unsigned N = (unsigned)(in_sizes[0] / IN);   // 50000
    const unsigned E = (unsigned)(in_sizes[1] / 2);    // 800000

    const int* src = edge;
    const int* dst = edge + E;

    const int NT1 = HID / 16;                     // 4
    const int NT2 = (OUT + 15) / 16;              // 3 (padded to 48)
    const int numTiles = (int)(N / 16);           // 3125 (N divisible by 16)

    // workspace carve-out
    char* ws = (char*)d_ws;
    size_t off = 0;
    auto carve = [&](size_t bytes) -> void* {
        void* p = ws + off;
        off = (off + bytes + 255) & ~(size_t)255;
        return p;
    };
    float*    dinv  = (float*)   carve((size_t)N * 4);            // deg -> rsqrt(deg)
    _Float16* xh    = (_Float16*)carve((size_t)N * IN * 2);       // node_feat in f16
    _Float16* W1b   = (_Float16*)carve((size_t)(IN/32) * NT1 * 512 * 2);
    float*    h1    = (float*)   carve((size_t)N * HID * 4);      // x @ W1
    float*    out1  = (float*)   carve((size_t)N * HID * 4);      // aggregated layer-1
    float*    sums  = (float*)   carve(128 * 4);                  // BN sum / sumsq
    _Float16* x2h   = (_Float16*)carve((size_t)N * HID * 2);      // post-BN/ReLU f16
    _Float16* W2b   = (_Float16*)carve((size_t)(HID/32) * NT2 * 512 * 2);
    float*    h2    = (float*)   carve((size_t)N * OUT * 4);      // x2 @ W2
    float*    out   = (float*)d_out;
    (void)ws_size; (void)n_in; (void)out_size;

    const unsigned TB = 256;

    // degrees (self-loop contributes 1) -> dinv
    fill_f32<<<cdiv(N, TB), TB, 0, stream>>>(dinv, 1.0f, N);
    deg_count<<<cdiv(E, TB), TB, 0, stream>>>(dst, dinv, E);
    deg_to_dinv<<<cdiv(N, TB), TB, 0, stream>>>(dinv, N);

    // ---- layer 1: h1 = node_feat @ W1 (WMMA f16 -> f32) ----
    f32_to_f16<<<cdiv((size_t)N * IN, TB), TB, 0, stream>>>(node_feat, xh, N * (unsigned)IN);
    pack_B<<<cdiv((size_t)(IN/32) * NT1 * 512, TB), TB, 0, stream>>>(W1, W1b, IN, HID, NT1);
    gemm_wmma<4><<<cdiv((size_t)numTiles * 32, TB), TB, 0, stream>>>(xh, W1b, h1, IN, numTiles, HID);

    // ---- aggregate 1 ----
    fill_f32<<<cdiv((size_t)N * HID, TB), TB, 0, stream>>>(out1, 0.0f, N * (unsigned)HID);
    scatter_edges<<<cdiv((size_t)E * HID, TB), TB, 0, stream>>>(src, dst, dinv, h1, out1, E, HID);
    post_agg<<<cdiv((size_t)N * HID, TB), TB, 0, stream>>>(h1, dinv, b1, out1, N, HID);

    // ---- batchnorm + relu -> f16 ----
    fill_f32<<<1, 128, 0, stream>>>(sums, 0.0f, 128);
    bn_stats<<<256, TB, 0, stream>>>(out1, sums, N);
    bn_relu_f16<<<cdiv((size_t)N * HID, TB), TB, 0, stream>>>(out1, sums, gamma, beta, x2h, N);

    // ---- layer 2: h2 = x2 @ W2 (WMMA, cols padded 40->48) ----
    pack_B<<<cdiv((size_t)(HID/32) * NT2 * 512, TB), TB, 0, stream>>>(W2, W2b, HID, OUT, NT2);
    gemm_wmma<3><<<cdiv((size_t)numTiles * 32, TB), TB, 0, stream>>>(x2h, W2b, h2, HID, numTiles, OUT);

    // ---- aggregate 2 -> d_out ----
    fill_f32<<<cdiv((size_t)N * OUT, TB), TB, 0, stream>>>(out, 0.0f, N * (unsigned)OUT);
    scatter_edges<<<cdiv((size_t)E * OUT, TB), TB, 0, stream>>>(src, dst, dinv, h2, out, E, OUT);
    post_agg<<<cdiv((size_t)N * OUT, TB), TB, 0, stream>>>(h2, dinv, b2, out, N, OUT);
}
